// CAC_42511586296007
// MI455X (gfx1250) — matrix operand order repaired and checked by reading, not compiled, not run on hardware
//
#include <hip/hip_runtime.h>
#include <stdint.h>

#define B_    32
#define C_    1024
#define HW_   784
#define NM_   154            // 1 global + 153 masked descriptors
#define ROWS_ (NM_ * B_)     // 4928 (m,b) rows in the VT scratch matrix

typedef _Float16 v16h __attribute__((ext_vector_type(16)));
typedef float    v8f  __attribute__((ext_vector_type(8)));

// ---------------- stage 1: one-pass ring/quadrant binning ----------------
// grid = 32 (b) * 128 (c-groups), block = 256 threads = 8 waves, 1 wave per channel.
__global__ __launch_bounds__(256) void cac_stage1(const float* __restrict__ x,
                                                  float* __restrict__ vt) {
  __shared__ __align__(16) float tile[8][800];   // 784 used per wave, padded
  __shared__ unsigned binsu[8][80];              // 77 bins used per wave

  const int w    = threadIdx.x >> 5;
  const int lane = threadIdx.x & 31;
  const int b    = blockIdx.x >> 7;
  const int cg   = blockIdx.x & 127;
  const int c    = cg * 8 + w;

  for (int j = lane; j < 80; j += 32) binsu[w][j] = 0u;

  // Async-stage the 28x28 map (784 floats = 196 x 16B) into LDS.
  const float* gbase = x + (size_t)(b * C_ + c) * HW_;
  for (int i = 0; i < 7; ++i) {
    int idx = i * 32 + lane;
    if (idx < 196) {
      uint32_t loff = (uint32_t)(size_t)(&tile[w][idx * 4]);
      asm volatile("global_load_async_to_lds_b128 %0, %1, off"
                   :: "v"(loff), "v"(gbase + idx * 4) : "memory");
    }
  }
  asm volatile("s_wait_asynccnt 0" ::: "memory");

  // Bin pixels: each pixel is in exactly one ring, 1-4 quadrants.
  float gm = -3.4e38f;
  for (int p = lane; p < HW_; p += 32) {
    float val = tile[w][p];
    gm = fmaxf(gm, val);
    int row = p / 28, col = p - row * 28;
    int yv = row - 14, xv = col - 14;       // y = first index, x = second (per reference)
    int r2 = xv * xv + yv * yv;
    unsigned bits = __float_as_uint(fmaxf(val, 0.0f));   // masks always contain zeros
    if (r2 == 0) {
      binsu[w][0] = bits;                   // center mask: exactly one pixel
    } else {
      int r = (int)sqrtf((float)r2);
      while (r * r < r2) ++r;               // smallest r with (r-1)^2 < r2 <= r^2
      int base = 1 + (r - 1) * 4;
      if (xv >= 0 && yv >= 0) atomicMax(&binsu[w][base + 0], bits);
      if (xv <= 0 && yv >= 0) atomicMax(&binsu[w][base + 1], bits);
      if (xv >= 0 && yv <= 0) atomicMax(&binsu[w][base + 2], bits);
      if (xv <= 0 && yv <= 0) atomicMax(&binsu[w][base + 3], bits);
    }
  }
  // wave32 reduction for the unmasked global max
  for (int m = 16; m; m >>= 1) gm = fmaxf(gm, __shfl_xor(gm, m, 32));
  __syncthreads();

  const float bin0 = __uint_as_float(binsu[w][0]);
  if (lane == 0) {
    vt[(size_t)(0 * B_ + b) * C_ + c] = gm;     // m=0: global (true) max
    vt[(size_t)(1 * B_ + b) * C_ + c] = bin0;   // m=1: center mask
  }
  if (lane < 4) {                                // one lane per quadrant
    int q = lane;
    float pre = bin0;                            // circles include the center
    for (int r = 1; r <= 19; ++r) {
      float rv = __uint_as_float(binsu[w][1 + (r - 1) * 4 + q]);
      int mring = 2 + (r - 1) * 8 + q * 2;
      vt[(size_t)(mring * B_ + b) * C_ + c] = rv;          // ring mask
      pre = fmaxf(pre, rv);
      vt[(size_t)((mring + 1) * B_ + b) * C_ + c] = pre;   // circle mask (prefix max)
    }
  }
}

// ---------------- stage 2: L2 norms = Gram diagonal via WMMA ----------------
// grid = 308 blocks of 16 rows; block = 8 waves, each wave covers 4 K-chunks of 32.
__global__ __launch_bounds__(256) void cac_stage2(const float* __restrict__ vt,
                                                  float* __restrict__ scale) {
  __shared__ float chunk[8][16][33];
  __shared__ float dbuf[8][32][8];
  const int w = threadIdx.x >> 5, lane = threadIdx.x & 31;
  const int r0 = blockIdx.x * 16;
  const int M = lane & 15, hi = lane >> 4;

  v8f acc = {};
  for (int kc = w; kc < 32; kc += 8) {
    for (int i = 0; i < 16; ++i)   // coalesced stage of the 16x32 f32 chunk
      chunk[w][i][lane] = vt[(size_t)(r0 + i) * C_ + kc * 32 + lane];
    // same-wave LDS producer->consumer is in-order; build fragments
    v16h a, bm;
#pragma unroll
    for (int e = 0; e < 16; ++e) {
      int k = (e < 8 ? e : e + 8) + hi * 8;          // 16-bit A-matrix layout
      a[e]  = (_Float16)chunk[w][M][k];              // A[M][K]
      bm[e] = (_Float16)chunk[w][M][k];              // B[K][N=M] = A^T
    }
    acc = __builtin_amdgcn_wmma_f32_16x16x32_f16(false, a, false, bm,
                                                 (short)0, acc, false, false);
  }
  // spill per-wave Gram tiles; sum diagonals in fixed order (deterministic)
#pragma unroll
  for (int r = 0; r < 8; ++r) dbuf[w][lane][r] = acc[r];
  __syncthreads();
  if (threadIdx.x < 16) {
    int i = threadIdx.x;
    int sl = (i < 8) ? i : (i + 16);       // lane holding D[i][i]
    float s = 0.0f;
    for (int ww = 0; ww < 8; ++ww) s += dbuf[ww][sl][i & 7];
    scale[r0 + i] = 1.0f / (sqrtf(s) + 1e-6f);
  }
}

// ---------------- stage 3: weighted accumulation over descriptors ----------------
__global__ __launch_bounds__(256) void cac_stage3(const float* __restrict__ vt,
                                                  const float* __restrict__ scale,
                                                  float* __restrict__ out) {
  int idx = blockIdx.x * 256 + threadIdx.x;   // 0..32767
  int b = idx >> 10, c = idx & 1023;
  float acc = 0.0f;
#pragma unroll 7
  for (int m = 0; m < NM_; ++m) {
    int p = m * B_ + b;
    acc += vt[(size_t)p * C_ + c] * scale[p];
  }
  out[idx] = acc;
}

extern "C" void kernel_launch(void* const* d_in, const int* in_sizes, int n_in,
                              void* d_out, int out_size, void* d_ws, size_t ws_size,
                              hipStream_t stream) {
  (void)in_sizes; (void)n_in; (void)out_size; (void)ws_size;
  const float* x   = (const float*)d_in[0];
  float* out       = (float*)d_out;
  float* vt        = (float*)d_ws;                 // ROWS_ x C_ floats (~20.2 MB)
  float* scale     = vt + (size_t)ROWS_ * C_;      // ROWS_ floats

  cac_stage1<<<dim3(B_ * 128), dim3(256), 0, stream>>>(x, vt);
  cac_stage2<<<dim3(ROWS_ / 16), dim3(256), 0, stream>>>(vt, scale);
  cac_stage3<<<dim3((B_ * C_) / 256), dim3(256), 0, stream>>>(vt, scale, out);
}
